// CrossChannelAttention_37452114821537
// MI455X (gfx1250) — matrix operand order repaired and checked by reading, not compiled, not run on hardware
//
#include <hip/hip_runtime.h>
#include <hip/hip_bf16.h>

// ---------------- problem constants ----------------
#define Bn   2
#define Cch  64
#define Hh   96
#define Wwd  96
#define Nn   (Hh*Wwd)     // 9216 tokens
#define NFt  192          // value / channel dim
#define RDt  24           // q/k dim
#define RDP  32           // padded q/k dim for K=32 WMMA
#define BM   64           // query rows per workgroup (4 waves x 16)
#define BN   64           // keys per inner tile
#define FT   12           // 192/16 feature tiles

#define USE_ASYNC_LDS 1   // CDNA5 GLOBAL_LOAD_ASYNC_TO_LDS path (ASYNCcnt)

typedef _Float16 v16h __attribute__((ext_vector_type(16)));
typedef _Float16 v8h  __attribute__((ext_vector_type(8)));
typedef float    v8f  __attribute__((ext_vector_type(8)));

union V16 { v16h v; v8h h[2]; };

// ---------------- kernel 1: fused q/k/v 1x1 convs -> f16 ----------------
__global__ __launch_bounds__(256) void qkv_kernel(
    const float* __restrict__ r, const float* __restrict__ g, const float* __restrict__ b,
    const float* __restrict__ Wq, const float* __restrict__ bq,
    const float* __restrict__ Wk, const float* __restrict__ bk,
    const float* __restrict__ Wv, const float* __restrict__ bv,
    _Float16* __restrict__ Qh, _Float16* __restrict__ Kh, _Float16* __restrict__ Vh)
{
  __shared__ float rgbT[NFt][64];
  const int nb   = Nn / 64;
  const int bidx = blockIdx.x / nb;
  const int n0   = (blockIdx.x % nb) * 64;
  const int tid  = threadIdx.x;

  for (int i = tid; i < NFt * 64; i += 256) {
    int c = i >> 6, n = i & 63;
    const float* src = (c < Cch) ? r : (c < 2 * Cch) ? g : b;
    rgbT[c][n] = src[((size_t)bidx * Cch + (c & 63)) * Nn + n0 + n];
  }
  __syncthreads();

  const int n  = tid & 63;
  const int og = tid >> 6;                 // wave-uniform output group
  for (int oo = 0; oo < 60; ++oo) {
    int o = og * 60 + oo;                  // 0..239 : q(24) | k(24) | v(192)
    const float* Wrow; float bias; _Float16* dst;
    if (o < RDt)      { Wrow = Wq + (size_t)o * NFt;          bias = bq[o];
                        dst = Qh + ((size_t)bidx * Nn + n0 + n) * RDP + o; }
    else if (o < 48)  { Wrow = Wk + (size_t)(o - RDt) * NFt;  bias = bk[o - RDt];
                        dst = Kh + ((size_t)bidx * Nn + n0 + n) * RDP + (o - RDt); }
    else              { Wrow = Wv + (size_t)(o - 48) * NFt;   bias = bv[o - 48];
                        dst = Vh + ((size_t)bidx * NFt + (o - 48)) * Nn + n0 + n; }
    float acc = bias;
    #pragma unroll 8
    for (int c = 0; c < NFt; ++c) acc += Wrow[c] * rgbT[c][n];
    *dst = (_Float16)acc;
  }
  // zero the padded features 24..31 of Q and K
  if (tid < 64) {
    v8h z = {};
    *(v8h*)(Qh + ((size_t)bidx * Nn + n0 + tid) * RDP + 24) = z;
    *(v8h*)(Kh + ((size_t)bidx * Nn + n0 + tid) * RDP + 24) = z;
  }
}

// ---------------- async global->LDS helpers (CDNA5 TDM-adjacent path) ----------------
__device__ __forceinline__ void async_fence() {
#if USE_ASYNC_LDS
#if __has_builtin(__builtin_amdgcn_s_wait_asynccnt)
  __builtin_amdgcn_s_wait_asynccnt(0);
#else
  asm volatile("s_wait_asynccnt 0x0" ::: "memory");
#endif
#endif
}

__device__ __forceinline__ void copy16_g2l(void* ldsDst, const void* gSrc) {
#if USE_ASYNC_LDS
  unsigned lds = (unsigned)(uintptr_t)ldsDst;   // low 32 bits of generic = LDS byte offset
  asm volatile("global_load_async_to_lds_b128 %0, %1, off"
               :: "v"(lds), "v"(gSrc) : "memory");
#else
  *(v8h*)ldsDst = *(const v8h*)gSrc;
#endif
}

// ---------------- kernel 2: flash attention + output residual ----------------
// LDS layout (bytes), double-buffered K/V:
#define VSTRIDE 72        // V^T row stride in halfs (bank-conflict padding)
#define SSTRIDE 66        // S row stride in floats
#define OSTRIDE 197       // O staging row stride in floats
#define OFF_K0  0                              // 64*32*2   = 4096
#define OFF_K1  4096                           // 4096
#define OFF_V0  8192                           // 192*72*2  = 27648
#define OFF_V1  35840                          // 27648  -> 63488
#define OFF_S   63488                          // 4*16*66*4 = 16896 -> 80384
#define OFF_P   80384                          // 4*16*64*2 = 8192  -> 88576
#define OFF_RA  88576                          // 4*16*4    = 256
#define OFF_RL  88832                          // 4*16*4    = 256
#define SMEM_TOTAL 89088                       // O staging (64*197*4=50432) reuses [0,..)

__device__ __forceinline__ void stage_tile(const _Float16* __restrict__ Kbase,
                                           const _Float16* __restrict__ Vbase,
                                           int kb, _Float16* kdst, _Float16* vdst, int tid)
{
  // K tile: 64 keys x 32 halfs (row-major by key), 256 x 16B chunks
  for (int i = tid; i < (BN * RDP) / 8; i += 128) {
    int key = i >> 2, part = i & 3;
    copy16_g2l(kdst + key * RDP + part * 8,
               Kbase + (size_t)(kb + key) * RDP + part * 8);
  }
  // V tile transposed: vdst[feature][key]; coalesced since V stored (NF,N)
  for (int i = tid; i < (NFt * BN) / 8; i += 128) {
    int f = i >> 3, part = i & 7;
    copy16_g2l(vdst + f * VSTRIDE + part * 8,
               Vbase + (size_t)f * Nn + kb + part * 8);
  }
}

__global__ __launch_bounds__(128) void flash_kernel(
    const _Float16* __restrict__ Qh, const _Float16* __restrict__ Kh,
    const _Float16* __restrict__ Vh,
    const float* __restrict__ rin, const float* __restrict__ gin,
    const float* __restrict__ bin, float* __restrict__ out)
{
  __shared__ __align__(16) char smem[SMEM_TOTAL];
  _Float16* kbuf0 = (_Float16*)(smem + OFF_K0);
  _Float16* kbuf1 = (_Float16*)(smem + OFF_K1);
  _Float16* vbuf0 = (_Float16*)(smem + OFF_V0);
  _Float16* vbuf1 = (_Float16*)(smem + OFF_V1);
  float*    Slds = (float*)   (smem + OFF_S);   // [4][16][SSTRIDE]
  _Float16* Plds = (_Float16*)(smem + OFF_P);   // [4][16][BN]
  float*    rowA = (float*)   (smem + OFF_RA);  // [4][16] alpha broadcast
  float*    rowL = (float*)   (smem + OFF_RL);  // [4][16] row sums
  float*    Olds = (float*)   (smem);           // [BM][OSTRIDE] (reuse after loop)

  const int nb   = Nn / BM;
  const int bidx = blockIdx.x / nb;
  const int j0   = (blockIdx.x % nb) * BM;
  const int tid  = threadIdx.x;
  const int wave = tid >> 5, lane = tid & 31;
  const int m = lane & 15, hi = lane >> 4;

  // Q A-fragment (16x32 f16): 16 halves = feats hi*16..hi*16+15
  const _Float16* qp = Qh + ((size_t)bidx * Nn + j0 + wave * 16 + m) * RDP;
  V16 qa;
  qa.h[0] = *(const v8h*)(qp + hi * 16);
  qa.h[1] = *(const v8h*)(qp + hi * 16 + 8);

  v8f o[FT];
  #pragma unroll
  for (int f = 0; f < FT; ++f) o[f] = v8f{};

  float mrow = -1e30f, lrow = 0.f;
  float*    Sw = Slds + wave * 16 * SSTRIDE;
  _Float16* Pw = Plds + wave * 16 * BN;
  const _Float16* Kbase = Kh + (size_t)bidx * Nn * RDP;
  const _Float16* Vbase = Vh + (size_t)bidx * NFt * Nn;

  // prologue: async-issue tile 0 into buffer 0
  stage_tile(Kbase, Vbase, 0, kbuf0, vbuf0, tid);

  for (int kb = 0; kb < Nn; kb += BN) {
    const int cur = (kb / BN) & 1;
    _Float16* Klds = cur ? kbuf1 : kbuf0;
    _Float16* Vlds = cur ? vbuf1 : vbuf0;

    async_fence();                              // current tile landed in LDS
    __syncthreads();                            // ..for every wave; prev buffer free
    if (kb + BN < Nn)                           // prefetch next tile into other buffer
      stage_tile(Kbase, Vbase, kb + BN, cur ? kbuf0 : kbuf1, cur ? vbuf0 : vbuf1, tid);

    // ---- S = Q K^T  (16 x 64 per wave), 4 WMMAs ----
    #pragma unroll
    for (int nt = 0; nt < 4; ++nt) {
      V16 kf;   // B-frag 32x16: lane n = key nt*16+m, halves = feats hi*16..+15
      const _Float16* kp = Klds + (nt * 16 + m) * RDP + hi * 16;
      kf.h[0] = *(const v8h*)kp;
      kf.h[1] = *(const v8h*)(kp + 8);
      v8f c = v8f{};
      c = __builtin_amdgcn_wmma_f32_16x16x32_f16(false, qa.v, false, kf.v,
                                                 (short)0, c, false, false);
      #pragma unroll
      for (int i = 0; i < 8; ++i)               // C layout: row=(hi*8+i), col=nt*16+m
        Sw[(hi * 8 + i) * SSTRIDE + nt * 16 + m] = c[i];
    }

    // ---- online softmax: 2 lanes per row (cols split 0..31 / 32..63) ----
    {
      const int c0 = hi * 32;
      float mx = -1e30f;
      #pragma unroll 8
      for (int j = 0; j < 32; ++j) mx = fmaxf(mx, Sw[m * SSTRIDE + c0 + j]);
      mx = fmaxf(mx, __shfl_xor(mx, 16, 32));
      float mnew  = fmaxf(mrow, mx);
      float alpha = __expf(mrow - mnew);
      float ls = 0.f;
      #pragma unroll 8
      for (int j = 0; j < 32; ++j) {
        float p = __expf(Sw[m * SSTRIDE + c0 + j] - mnew);
        Pw[m * BN + c0 + j] = (_Float16)p;
        ls += p;
      }
      ls += __shfl_xor(ls, 16, 32);
      lrow = lrow * alpha + ls;
      mrow = mnew;
      if (hi == 0) rowA[wave * 16 + m] = alpha;
    }

    // rescale O accumulators by per-row alpha (broadcast via LDS)
    #pragma unroll
    for (int i = 0; i < 8; ++i) {
      float a = rowA[wave * 16 + hi * 8 + i];
      #pragma unroll
      for (int f = 0; f < FT; ++f) o[f][i] *= a;
    }

    // ---- O += P V : 2 A-frags (keys 0..31 / 32..63), 12 feature tiles ----
    V16 pa0, pa1;
    pa0.h[0] = *(const v8h*)(Pw + m * BN + hi * 16);
    pa0.h[1] = *(const v8h*)(Pw + m * BN + hi * 16 + 8);
    pa1.h[0] = *(const v8h*)(Pw + m * BN + 32 + hi * 16);
    pa1.h[1] = *(const v8h*)(Pw + m * BN + 32 + hi * 16 + 8);
    #pragma unroll
    for (int f = 0; f < FT; ++f) {
      const _Float16* vp = Vlds + (f * 16 + m) * VSTRIDE;   // lane n = feature f*16+m
      V16 vb0, vb1;                                         // halves = keys hi*16..+15
      vb0.h[0] = *(const v8h*)(vp + hi * 16);
      vb0.h[1] = *(const v8h*)(vp + hi * 16 + 8);
      vb1.h[0] = *(const v8h*)(vp + 32 + hi * 16);
      vb1.h[1] = *(const v8h*)(vp + 32 + hi * 16 + 8);
      o[f] = __builtin_amdgcn_wmma_f32_16x16x32_f16(false, pa0.v, false, vb0.v,
                                                    (short)0, o[f], false, false);
      o[f] = __builtin_amdgcn_wmma_f32_16x16x32_f16(false, pa1.v, false, vb1.v,
                                                    (short)0, o[f], false, false);
    }
  }

  // normalize rows by softmax sum
  if (hi == 0) rowL[wave * 16 + m] = lrow;
  #pragma unroll
  for (int i = 0; i < 8; ++i) {
    float inv = 1.f / rowL[wave * 16 + hi * 8 + i];
    #pragma unroll
    for (int f = 0; f < FT; ++f) o[f][i] *= inv;
  }

  __syncthreads();                              // all buffers dead; reuse base for O
  // stage O (transpose through LDS for coalesced global writes)
  #pragma unroll
  for (int f = 0; f < FT; ++f)
    #pragma unroll
    for (int i = 0; i < 8; ++i)
      Olds[(wave * 16 + hi * 8 + i) * OSTRIDE + f * 16 + m] = o[f][i];
  __syncthreads();

  // out tuple layout: [3][B][C][N]; add residual input
  for (int i = tid; i < BM * NFt; i += 128) {
    int c = i >> 6, jj = i & 63;                // c: 0..191, jj: query-row offset
    int t = c >> 6, cl = c & 63;
    const float* src = (t == 0) ? rin : (t == 1) ? gin : bin;
    size_t gi = ((size_t)bidx * Cch + cl) * Nn + j0 + jj;
    size_t go = (((size_t)t * Bn + bidx) * Cch + cl) * Nn + j0 + jj;
    out[go] = src[gi] + Olds[jj * OSTRIDE + c];
  }
}

// ---------------- host launcher ----------------
extern "C" void kernel_launch(void* const* d_in, const int* in_sizes, int n_in,
                              void* d_out, int out_size, void* d_ws, size_t ws_size,
                              hipStream_t stream) {
  (void)in_sizes; (void)n_in; (void)out_size; (void)ws_size;
  const float* r  = (const float*)d_in[0];
  const float* g  = (const float*)d_in[1];
  const float* b  = (const float*)d_in[2];
  const float* Wq = (const float*)d_in[3];
  const float* bq = (const float*)d_in[4];
  const float* Wk = (const float*)d_in[5];
  const float* bk = (const float*)d_in[6];
  const float* Wv = (const float*)d_in[7];
  const float* bv = (const float*)d_in[8];
  float* out = (float*)d_out;

  // workspace: Qh (B,N,32) f16 | Kh (B,N,32) f16 | Vh (B,192,N) f16  (~9.4 MB)
  _Float16* Qh = (_Float16*)d_ws;
  _Float16* Kh = Qh + (size_t)Bn * Nn * RDP;
  _Float16* Vh = Kh + (size_t)Bn * Nn * RDP;

  qkv_kernel<<<Bn * (Nn / 64), 256, 0, stream>>>(r, g, b, Wq, bq, Wk, bk, Wv, bv,
                                                 Qh, Kh, Vh);
  flash_kernel<<<Bn * (Nn / BM), 128, 0, stream>>>(Qh, Kh, Vh, r, g, b, out);
}
